// DioraBase_72541997629721
// MI455X (gfx1250) — matrix operand order repaired and checked by reading, not compiled, not run on hardware
//
#include <hip/hip_runtime.h>
#include <hip/hip_bf16.h>
#include <math.h>

// ---------------- problem constants ----------------
#define Bb     32
#define Tt     40
#define DIN    1024
#define Dd     400
#define NCELLS (Tt * (Tt + 1) / 2)   // 820
#define NT     25                    // Dd / 16 column tiles
#define OUTW   (2 * Dd + 2)          // 802

typedef __attribute__((ext_vector_type(16))) __bf16 v16bf;
typedef __attribute__((ext_vector_type(8)))  float  v8f;
typedef __bf16 bf16;

__device__ __forceinline__ int offc(int lev) { return lev * Tt - lev * (lev - 1) / 2; }

// ---------------- small utility kernels ----------------
__global__ void k_zero(float* p, int n) {
    int i = blockIdx.x * 256 + threadIdx.x;
    if (i < n) p[i] = 0.f;
}

__global__ void k_f32_to_bf16(const float* __restrict__ s, bf16* __restrict__ d, int n) {
    int i = blockIdx.x * 256 + threadIdx.x;
    if (i < n) d[i] = (bf16)s[i];
}

__global__ void k_iota(int* p, int n) {
    int i = blockIdx.x * 256 + threadIdx.x;
    if (i < n) p[i] = i;
}

// Pack a K x 400 f32 weight matrix into WMMA B-fragment order (bf16):
// fragment element ((kt*NT + nt)*32 + lane)*16 + h  holds  W[kt*32 + kk(h, lane/16)][nt*16 + lane%16]
// with kk = 8*(lane/16) + h for h < 8, 16 + 8*(lane/16) + (h-8) for h >= 8. Zero-pad k >= Kdim.
__global__ void k_pack_w(const float* __restrict__ W, bf16* __restrict__ dst, int Kdim, int KT) {
    int t = blockIdx.x * 256 + threadIdx.x;
    int total = KT * NT * 32;
    if (t >= total) return;
    int lane = t & 31;
    int rest = t >> 5;
    int nt = rest % NT;
    int kt = rest / NT;
    int hg = lane >> 4;
    bf16* o = dst + (size_t)t * 16;
    int cbase = nt * 16 + (lane & 15);
    #pragma unroll
    for (int h = 0; h < 16; ++h) {
        int k = kt * 32 + (h < 8 ? 8 * hg + h : 16 + 8 * hg + (h - 8));
        float v = (k < Kdim) ? W[(size_t)k * Dd + cbase] : 0.f;
        o[h] = (bf16)v;
    }
}

// ---------------- chart index generation ----------------
__global__ void k_idx_inside(int* __restrict__ idxL, int* __restrict__ idxR, int level) {
    int L = Tt - level;
    int M = Bb * L * level;
    int m = blockIdx.x * 256 + threadIdx.x;
    if (m >= M) return;
    int per = L * level;
    int b = m / per;
    int rem = m - b * per;
    int pos = rem / level;
    int k = rem - pos * level;
    int li = offc(k) + pos;
    int ri = offc(level - k - 1) + pos + k + 1;
    idxL[m] = b * NCELLS + li;
    idxR[m] = b * NCELLS + ri;
}

__global__ void k_idx_outside(int* __restrict__ idxL, int* __restrict__ idxR, int level) {
    int L = Tt - level;
    int N = L - 1;
    int M = Bb * L * N;
    int m = blockIdx.x * 256 + threadIdx.x;
    if (m >= M) return;
    int per = L * N;
    int b = m / per;
    int rem = m - b * per;
    int pos = rem / N;
    int j = rem - pos * N;
    int pi, si;
    if (j < pos) {
        pi = offc(level + j + 1) + pos - j - 1;
        si = offc(j) + pos - j - 1;
    } else {
        int j2 = j - pos;
        pi = offc(level + j2 + 1) + pos;
        si = offc(j2) + pos + level + 1;
    }
    idxL[m] = b * NCELLS + pi;  // h_out parent
    idxR[m] = b * NCELLS + si;  // h_in sibling
}

// ---------------- WMMA GEMM ----------------
// out[M x 400] = act( gather([srcL|srcR]) @ Wpacked + bias )
// One block owns a full 16-row x 400-col output stripe:
//   - 128 threads stage the 16 x KP A tile into LDS with b128 gathers (once per stripe)
//   - each of the 4 waves sweeps PAIRS of column tiles (nt0, nt0+1), sharing one
//     ds_load_b128 A fragment between two v_wmma_f32_16x16x32_bf16 issues
//     (2 ds + 4 vmem b128 per 2 WMMAs). NT=25 -> last tile takes the single path.
__global__ __launch_bounds__(128) void k_gemm_wmma(
    const bf16* __restrict__ srcL, const int* __restrict__ idxL, int SL, int KL,
    const bf16* __restrict__ srcR, const int* __restrict__ idxR, int SR, int KR,
    int KP, const bf16* __restrict__ Wp,
    const float* __restrict__ bias, int relu,
    float* __restrict__ outp, int M)
{
    __shared__ __align__(16) bf16 At[16][1024 + 8];  // row stride 1032 halves (16B aligned)
    const int tid = threadIdx.y * 32 + threadIdx.x;
    const int m0 = blockIdx.x * 16;

    // cooperative vectorized gather of the 16 x KP A tile (uint4 = 8 bf16 per load)
    const int CL = KL >> 3;          // chunks sourced from srcL
    const int CK = (KL + KR) >> 3;   // total data chunks
    const int CP = KP >> 3;          // padded chunks per row
    for (int t = tid; t < 16 * CP; t += 128) {
        int r = t / CP;
        int c = t - r * CP;
        uint4 v = make_uint4(0u, 0u, 0u, 0u);
        int m = m0 + r;
        if (m < M) {
            if (c < CL)
                v = *(const uint4*)(srcL + (size_t)idxL[m] * SL + (c << 3));
            else if (c < CK)
                v = *(const uint4*)(srcR + (size_t)idxR[m] * SR + ((c - CL) << 3));
        }
        *(uint4*)(&At[r][c << 3]) = v;
    }
    __syncthreads();

    const int lane = threadIdx.x;
    const int g = lane >> 4;        // A lane group (K half-select)
    const int rr = lane & 15;       // A row / C column
    const int KT = KP >> 5;

    auto epilogue = [&](int nt, v8f acc) {
        int c = nt * 16 + rr;
        float bv = bias ? bias[c] : 0.f;
        #pragma unroll
        for (int v = 0; v < 8; ++v) {
            int m = m0 + v + 8 * g;  // C: VGPR v holds row v (+8 for upper lane half)
            if (m < M) {
                float x = acc[v] + bv;
                if (relu) x = x > 0.f ? x : 0.f;
                outp[(size_t)m * Dd + c] = x;
            }
        }
    };

    for (int nt0 = threadIdx.y * 2; nt0 < NT; nt0 += 8) {
        if (nt0 + 1 < NT) {
            // dual-accumulator path: one A fragment feeds two WMMAs
            v8f acc0 = {}, acc1 = {};
            for (int kt = 0; kt < KT; ++kt) {
                int kk = kt << 5;
                union { uint4 u[2]; v16bf v; } a, b0, b1;
                const bf16* ap = &At[rr][kk + g * 8];
                a.u[0] = *(const uint4*)(ap);
                a.u[1] = *(const uint4*)(ap + 16);
                const bf16* bp = Wp + ((((size_t)kt * NT + nt0) * 32 + lane) << 4);
                b0.u[0] = *(const uint4*)(bp);
                b0.u[1] = *(const uint4*)(bp + 8);
                b1.u[0] = *(const uint4*)(bp + 512);   // next nt: +32 lanes * 16 halves
                b1.u[1] = *(const uint4*)(bp + 520);
                acc0 = __builtin_amdgcn_wmma_f32_16x16x32_bf16(
                    false, a.v, false, b0.v, (short)0, acc0, false, false);
                acc1 = __builtin_amdgcn_wmma_f32_16x16x32_bf16(
                    false, a.v, false, b1.v, (short)0, acc1, false, false);
            }
            epilogue(nt0, acc0);
            epilogue(nt0 + 1, acc1);
        } else {
            v8f acc = {};
            for (int kt = 0; kt < KT; ++kt) {
                int kk = kt << 5;
                union { uint4 u[2]; v16bf v; } a, bfr;
                const bf16* ap = &At[rr][kk + g * 8];
                a.u[0] = *(const uint4*)(ap);
                a.u[1] = *(const uint4*)(ap + 16);
                const bf16* bp = Wp + ((((size_t)kt * NT + nt0) * 32 + lane) << 4);
                bfr.u[0] = *(const uint4*)(bp);
                bfr.u[1] = *(const uint4*)(bp + 8);
                acc = __builtin_amdgcn_wmma_f32_16x16x32_bf16(
                    false, a.v, false, bfr.v, (short)0, acc, false, false);
            }
            epilogue(nt0, acc);
        }
    }
}

// ---------------- bilinear finish: sb[m] = U[m].hR[m] + sL[idxL] + sR[idxR] ----------------
__global__ __launch_bounds__(256) void k_score(
    const float* __restrict__ U, const float* __restrict__ hR, const int* __restrict__ idxR,
    const float* __restrict__ sL, const int* __restrict__ idxL, const float* __restrict__ sR,
    float* __restrict__ sb, int M)
{
    int wid = threadIdx.x >> 5, lane = threadIdx.x & 31;
    int m = blockIdx.x * 8 + wid;
    if (m >= M) return;
    const float* u = U + (size_t)m * Dd;
    const float* h = hR + (size_t)idxR[m] * Dd;
    float acc = 0.f;
    for (int d = lane; d < Dd; d += 32) acc += u[d] * h[d];
    for (int off = 16; off > 0; off >>= 1) acc += __shfl_xor(acc, off, 32);
    if (lane == 0) sb[m] = acc + sL[idxL[m]] + sR[idxR[m]];
}

// ---------------- softmax over candidates + mix + unit-normalize ----------------
__global__ __launch_bounds__(256) void k_combine(
    const float* __restrict__ sb, const float* __restrict__ hc, int n, int L, int off,
    float* __restrict__ hdst, bf16* __restrict__ hdst_bf, float* __restrict__ sdst)
{
    __shared__ float p[64];
    __shared__ float red[256];
    __shared__ float scal[2];
    int b = blockIdx.y, pos = blockIdx.x, tid = threadIdx.x;
    size_t base = ((size_t)b * L + pos) * (size_t)n;
    if (tid == 0) {
        float mx = -3.4e38f;
        for (int k = 0; k < n; ++k) mx = fmaxf(mx, sb[base + k]);
        float s = 0.f;
        for (int k = 0; k < n; ++k) { float e = expf(sb[base + k] - mx); p[k] = e; s += e; }
        float inv = 1.f / s, sv = 0.f;
        for (int k = 0; k < n; ++k) { p[k] *= inv; sv += p[k] * sb[base + k]; }
        scal[0] = sv;
    }
    __syncthreads();
    int d0 = tid, d1 = tid + 256;
    float a0 = 0.f, a1 = 0.f;
    for (int k = 0; k < n; ++k) {
        const float* row = hc + (base + k) * Dd;
        float pk = p[k];
        a0 += pk * row[d0];
        if (d1 < Dd) a1 += pk * row[d1];
    }
    float ss = a0 * a0 + ((d1 < Dd) ? a1 * a1 : 0.f);
    red[tid] = ss;
    __syncthreads();
    for (int s = 128; s > 0; s >>= 1) { if (tid < s) red[tid] += red[tid + s]; __syncthreads(); }
    if (tid == 0) scal[1] = 1.f / (sqrtf(red[0]) + 1e-8f);
    __syncthreads();
    float inv = scal[1];
    size_t cell = (size_t)b * NCELLS + off + pos;
    float* hd = hdst + cell * Dd;
    bf16* hb = hdst_bf + cell * Dd;
    hd[d0] = a0 * inv; hb[d0] = (bf16)(a0 * inv);
    if (d1 < Dd) { hd[d1] = a1 * inv; hb[d1] = (bf16)(a1 * inv); }
    if (tid == 0) sdst[cell] = scal[0];
}

// ---------------- leaf normalize ----------------
__global__ __launch_bounds__(256) void k_leafnorm(
    const float* __restrict__ src, float* __restrict__ hdst, bf16* __restrict__ hdst_bf,
    int rowsPerB)
{
    __shared__ float red[256];
    __shared__ float sinv;
    int m = blockIdx.x, tid = threadIdx.x;
    const float* s = src + (size_t)m * Dd;
    float a0 = s[tid];
    float a1 = (tid + 256 < Dd) ? s[tid + 256] : 0.f;
    red[tid] = a0 * a0 + a1 * a1;
    __syncthreads();
    for (int st = 128; st > 0; st >>= 1) { if (tid < st) red[tid] += red[tid + st]; __syncthreads(); }
    if (tid == 0) sinv = 1.f / (sqrtf(red[0]) + 1e-8f);
    __syncthreads();
    int b = m / rowsPerB, t = m % rowsPerB;
    size_t cell = (size_t)b * NCELLS + t;
    float inv = sinv;
    hdst[cell * Dd + tid] = a0 * inv;
    hdst_bf[cell * Dd + tid] = (bf16)(a0 * inv);
    if (tid + 256 < Dd) {
        hdst[cell * Dd + tid + 256] = a1 * inv;
        hdst_bf[cell * Dd + tid + 256] = (bf16)(a1 * inv);
    }
}

// ---------------- root: unit(root_h) broadcast to every batch's root cell ----------------
__global__ __launch_bounds__(256) void k_root(
    const float* __restrict__ root_h, float* __restrict__ h_out, bf16* __restrict__ h_out_bf)
{
    __shared__ float red[256];
    __shared__ float sinv;
    int tid = threadIdx.x;
    float a0 = root_h[tid];
    float a1 = (tid + 256 < Dd) ? root_h[tid + 256] : 0.f;
    red[tid] = a0 * a0 + a1 * a1;
    __syncthreads();
    for (int st = 128; st > 0; st >>= 1) { if (tid < st) red[tid] += red[tid + st]; __syncthreads(); }
    if (tid == 0) sinv = 1.f / (sqrtf(red[0]) + 1e-8f);
    __syncthreads();
    float v0 = a0 * sinv, v1 = a1 * sinv;
    for (int b = 0; b < Bb; ++b) {
        size_t cell = (size_t)b * NCELLS + (NCELLS - 1);
        h_out[cell * Dd + tid] = v0;
        h_out_bf[cell * Dd + tid] = (bf16)v0;
        if (tid + 256 < Dd) {
            h_out[cell * Dd + tid + 256] = v1;
            h_out_bf[cell * Dd + tid + 256] = (bf16)v1;
        }
    }
}

// ---------------- final concat [h_in, s_in, h_out, s_out] ----------------
__global__ __launch_bounds__(256) void k_assemble(
    const float* __restrict__ h_in, const float* __restrict__ s_in,
    const float* __restrict__ h_out, const float* __restrict__ s_out,
    float* __restrict__ out)
{
    size_t r = blockIdx.x;
    int tid = threadIdx.x;
    float* o = out + r * OUTW;
    const float* hi = h_in + r * Dd;
    const float* ho = h_out + r * Dd;
    for (int d = tid; d < Dd; d += 256) { o[d] = hi[d]; o[Dd + 1 + d] = ho[d]; }
    if (tid == 0) { o[Dd] = s_in[r]; o[2 * Dd + 1] = s_out[r]; }
}

// ---------------- host orchestration ----------------
extern "C" void kernel_launch(void* const* d_in, const int* in_sizes, int n_in,
                              void* d_out, int out_size, void* d_ws, size_t ws_size,
                              hipStream_t stream)
{
    const float* x      = (const float*)d_in[0];
    const float* W_leaf = (const float*)d_in[1];
    const float* b_leaf = (const float*)d_in[2];
    const float* W_in   = (const float*)d_in[3];
    const float* b_in   = (const float*)d_in[4];
    const float* S_in   = (const float*)d_in[5];
    const float* W_out  = (const float*)d_in[6];
    const float* b_out  = (const float*)d_in[7];
    const float* S_out  = (const float*)d_in[8];
    const float* root_h = (const float*)d_in[9];

    char* ws = (char*)d_ws;
    size_t o = 0;
    auto alloc = [&](size_t bytes) -> char* {
        char* p = ws + o;
        o += (bytes + 255) & ~(size_t)255;
        return p;
    };

    const int maxM = Bb * Tt * (Tt - 1);  // 49920 (outside level 0)

    float* h_in_f   = (float*)alloc(sizeof(float) * Bb * NCELLS * Dd);
    float* h_out_f  = (float*)alloc(sizeof(float) * Bb * NCELLS * Dd);
    bf16*  h_in_b   = (bf16*) alloc(2ull * Bb * NCELLS * Dd);
    bf16*  h_out_b  = (bf16*) alloc(2ull * Bb * NCELLS * Dd);
    float* s_in_f   = (float*)alloc(sizeof(float) * Bb * NCELLS);
    float* s_out_f  = (float*)alloc(sizeof(float) * Bb * NCELLS);
    bf16*  x_b      = (bf16*) alloc(2ull * Bb * Tt * DIN);
    bf16*  Wleaf_p  = (bf16*) alloc(2ull * 32 * NT * 512);
    bf16*  Win_p    = (bf16*) alloc(2ull * 25 * NT * 512);
    bf16*  Sin_p    = (bf16*) alloc(2ull * 13 * NT * 512);
    bf16*  Wout_p   = (bf16*) alloc(2ull * 25 * NT * 512);
    bf16*  Sout_p   = (bf16*) alloc(2ull * 13 * NT * 512);
    int*   idxL     = (int*)  alloc(sizeof(int) * maxM);
    int*   idxR     = (int*)  alloc(sizeof(int) * maxM);
    float* hc       = (float*)alloc(sizeof(float) * (size_t)maxM * Dd);
    float* Ub       = (float*)alloc(sizeof(float) * (size_t)maxM * Dd);
    float* sbv      = (float*)alloc(sizeof(float) * maxM);

    const dim3 blkG(32, 4);

    // zero score charts (leaf s_in rows and root s_out stay zero)
    int nz = Bb * NCELLS;
    k_zero<<<(nz + 255) / 256, 256, 0, stream>>>(s_in_f, nz);
    k_zero<<<(nz + 255) / 256, 256, 0, stream>>>(s_out_f, nz);

    // bf16 conversions / weight packing
    int nx = Bb * Tt * DIN;
    k_f32_to_bf16<<<(nx + 255) / 256, 256, 0, stream>>>(x, x_b, nx);
    auto pack = [&](const float* W, bf16* dst, int Kdim, int KT) {
        int tot = KT * NT * 32;
        k_pack_w<<<(tot + 255) / 256, 256, 0, stream>>>(W, dst, Kdim, KT);
    };
    pack(W_leaf, Wleaf_p, 1024, 32);
    pack(W_in,   Win_p,   800,  25);
    pack(S_in,   Sin_p,   400,  13);
    pack(W_out,  Wout_p,  800,  25);
    pack(S_out,  Sout_p,  400,  13);

    // ---- leaves: h_in[:, :T] = unit(relu(x @ W_leaf + b_leaf)) ----
    int Ml = Bb * Tt;
    k_iota<<<(Ml + 255) / 256, 256, 0, stream>>>(idxL, Ml);
    k_gemm_wmma<<<dim3((Ml + 15) / 16), blkG, 0, stream>>>(
        x_b, idxL, DIN, DIN, nullptr, nullptr, 0, 0,
        1024, Wleaf_p, b_leaf, 1, hc, Ml);
    k_leafnorm<<<Ml, 256, 0, stream>>>(hc, h_in_f, h_in_b, Tt);

    // ---- root: h_out[:, -1] = unit(root_h) ----
    k_root<<<1, 256, 0, stream>>>(root_h, h_out_f, h_out_b);

    // ---- inside pass ----
    for (int level = 1; level < Tt; ++level) {
        int L = Tt - level;
        int M = Bb * L * level;
        k_idx_inside<<<(M + 255) / 256, 256, 0, stream>>>(idxL, idxR, level);
        dim3 gg((M + 15) / 16);
        // hc = relu([hl;hr] @ W_in + b_in)
        k_gemm_wmma<<<gg, blkG, 0, stream>>>(h_in_b, idxL, Dd, Dd,
                                             h_in_b, idxR, Dd, Dd,
                                             800, Win_p, b_in, 1, hc, M);
        // U = hl @ S_in
        k_gemm_wmma<<<gg, blkG, 0, stream>>>(h_in_b, idxL, Dd, Dd,
                                             nullptr, nullptr, 0, 0,
                                             416, Sin_p, nullptr, 0, Ub, M);
        // sb = U . hr + s_in[li] + s_in[ri]
        k_score<<<(M + 7) / 8, 256, 0, stream>>>(Ub, h_in_f, idxR,
                                                 s_in_f, idxL, s_in_f, sbv, M);
        int off = level * Tt - level * (level - 1) / 2;
        dim3 gc(L, Bb);
        k_combine<<<gc, 256, 0, stream>>>(sbv, hc, level, L, off,
                                          h_in_f, h_in_b, s_in_f);
    }

    // ---- outside pass ----
    for (int level = Tt - 2; level >= 0; --level) {
        int L = Tt - level;
        int N = L - 1;
        int M = Bb * L * N;
        k_idx_outside<<<(M + 255) / 256, 256, 0, stream>>>(idxL, idxR, level);
        dim3 gg((M + 15) / 16);
        // hc = relu([hp;hs] @ W_out + b_out)
        k_gemm_wmma<<<gg, blkG, 0, stream>>>(h_out_b, idxL, Dd, Dd,
                                             h_in_b, idxR, Dd, Dd,
                                             800, Wout_p, b_out, 1, hc, M);
        // U = hp @ S_out
        k_gemm_wmma<<<gg, blkG, 0, stream>>>(h_out_b, idxL, Dd, Dd,
                                             nullptr, nullptr, 0, 0,
                                             416, Sout_p, nullptr, 0, Ub, M);
        // sb = U . hs + s_out[pi] + s_in[si]
        k_score<<<(M + 7) / 8, 256, 0, stream>>>(Ub, h_in_f, idxR,
                                                 s_out_f, idxL, s_in_f, sbv, M);
        int off = level * Tt - level * (level - 1) / 2;
        dim3 gc(L, Bb);
        k_combine<<<gc, 256, 0, stream>>>(sbv, hc, N, L, off,
                                          h_out_f, h_out_b, s_out_f);
    }

    // ---- concat output ----
    k_assemble<<<Bb * NCELLS, 256, 0, stream>>>(h_in_f, s_in_f, h_out_f, s_out_f,
                                                (float*)d_out);
}
